// SSIM_8280696947190
// MI455X (gfx1250) — matrix operand order repaired and checked by reading, not compiled, not run on hardware
//
#include <hip/hip_runtime.h>
#include <hip/hip_bf16.h>

typedef __attribute__((ext_vector_type(16))) _Float16 v16h;
typedef __attribute__((ext_vector_type(8)))  float    v8f;

#define DEV __device__ __forceinline__

DEV v8f wmma_(v16h a, v16h b, v8f c) {
  // v_wmma_f32_16x16x32_f16 : D = A(16x32 f16) x B(32x16 f16) + C(16x16 f32)
  return __builtin_amdgcn_wmma_f32_16x16x32_f16(false, a, false, b, (short)0, c, false, false);
}

DEV float sigm_(float x) { return 1.0f / (1.0f + __expf(-x)); }
DEV float tanh_(float x) { float e = __expf(2.0f * x); return 1.0f - 2.0f / (e + 1.0f); }

// A fragment (16x32, f16) from a per-lane fp32 row pointer (row = this lane's M row).
// Layout: lane half h, VGPR v holds K pair k=2v+16*h (contiguous, matches B).
DEV v16h a_frag_row(const float* rowp, int k_base, int lane) {
  int half = (lane >> 4) * 16;
  v16h a;
#pragma unroll
  for (int v = 0; v < 8; ++v) {
    int k0 = 2 * v + half + k_base;
    a[2 * v]     = (_Float16)rowp[k0];
    a[2 * v + 1] = (_Float16)rowp[k0 + 1];
  }
  return a;
}

// A fragment from f16 LDS, row-major with leading dim ld.
DEV v16h a_frag_lds(const _Float16* base, int ld, int m_base, int k_base, int lane) {
  const _Float16* rowp = base + (m_base + (lane & 15)) * ld + k_base;
  int half = (lane >> 4) * 16;
  v16h a;
#pragma unroll
  for (int v = 0; v < 8; ++v) {
    int k0 = 2 * v + half;
    a[2 * v]     = rowp[k0];
    a[2 * v + 1] = rowp[k0 + 1];
  }
  return a;
}

// B fragment (32x16): B[k][n] = W[n*ld + k]; lanes 0-15 K=0..15, 16-31 K=16..31.
DEV v16h b_frag_w(const float* W, int ld, int n_base, int k_base, int lane) {
  const float* p = W + (n_base + (lane & 15)) * ld + k_base + ((lane >> 4) << 4);
  v16h b;
#pragma unroll
  for (int i = 0; i < 16; ++i) b[i] = (_Float16)p[i];
  return b;
}

// B fragment from a pre-converted f16 weight table (contiguous 32 B per lane).
DEV v16h b_frag_h16(const _Float16* W, int ld, int n_base, int k_base, int lane) {
  const _Float16* p = W + (n_base + (lane & 15)) * ld + k_base + ((lane >> 4) << 4);
  v16h b;
#pragma unroll
  for (int i = 0; i < 16; ++i) b[i] = p[i];
  return b;
}

// C/D fragment (16x16 f32) in normal row-major storage.
DEV v8f c_frag_load(const float* C, int ld, int m_base, int n_base, int lane) {
  int n  = n_base + (lane & 15);
  int mb = m_base + ((lane >> 4) << 3);
  v8f c;
#pragma unroll
  for (int r = 0; r < 8; ++r) c[r] = C[(mb + r) * ld + n];
  return c;
}

DEV void c_frag_store(float* C, int ld, int m_base, int n_base, int lane, v8f c) {
  int n  = n_base + (lane & 15);
  int mb = m_base + ((lane >> 4) << 3);
#pragma unroll
  for (int r = 0; r < 8; ++r) C[(mb + r) * ld + n] = c[r];
}

// Fragment-native 16x16 tile storage: element (lane, r) at p[lane*8 + r].
// Each lane touches one contiguous 32 B chunk -> b128-friendly on both sides.
DEV void c_tile_store(float* p, int lane, v8f c) {
  float* q = p + lane * 8;
#pragma unroll
  for (int r = 0; r < 8; ++r) q[r] = c[r];
}
DEV v8f c_tile_load(const float* p, int lane) {
  const float* q = p + lane * 8;
  v8f c;
#pragma unroll
  for (int r = 0; r < 8; ++r) c[r] = q[r];
  return c;
}

// ---------------------------------------------------------------------------
// K0: rev_idx[b*256+t] = (t < len[b]) ? len[b]-1-t : 0
__global__ void k_revidx(const int* __restrict__ len, int* __restrict__ rev) {
  int p = blockIdx.x * 256 + threadIdx.x;
  int b = p >> 8, t = p & 255;
  int L = len[b];
  rev[p] = (t < L) ? (L - 1 - t) : 0;
}

// K0b: one-time f32 -> f16 conversion of the decoder input-weight matrix.
__global__ void k_cvt(const float* __restrict__ src, _Float16* __restrict__ dst, int n) {
  int i = blockIdx.x * 256 + threadIdx.x;
  if (i < n) dst[i] = (_Float16)src[i];
}

// ---------------------------------------------------------------------------
// K1: input projections for both LSTM directions.
// Output in fragment-native tile order: tile index = mtile_global*32 + ntile,
// mtile_global = t*2 + (b>=16). The scan then streams contiguous 32 B/lane.
__global__ void k_xproj(const float* __restrict__ x,
                        const float* __restrict__ Wih_f, const float* __restrict__ b_f,
                        const float* __restrict__ Wih_b, const float* __restrict__ b_b,
                        const int* __restrict__ rev,
                        float* __restrict__ xpf, float* __restrict__ xpb) {
  int dir = blockIdx.y;
  const float* W    = dir ? Wih_b : Wih_f;
  const float* bias = dir ? b_b : b_f;
  float* xp         = dir ? xpb : xpf;
  int lane = threadIdx.x & 31, w = threadIdx.x >> 5;
  int mt = blockIdx.x;                      // 512 m-tiles of 16 rows (t*32+b order)
  int m = mt * 16 + (lane & 15);
  int t = m >> 5, b = m & 31;
  int st = dir ? rev[b * 256 + t] : t;      // backward scans time-reversed input
  const float* rowp = x + (b * 256 + st) * 64;
  v16h a0 = a_frag_row(rowp, 0, lane);
  v16h a1 = a_frag_row(rowp, 32, lane);
#pragma unroll
  for (int nt = 0; nt < 4; ++nt) {          // each wave: 4 of the 32 n-tiles
    int ntile = w * 4 + nt;
    int nb = ntile * 16;
    v16h bb0 = b_frag_w(W, 64, nb, 0, lane);
    v16h bb1 = b_frag_w(W, 64, nb, 32, lane);
    v8f c = {};
    c = wmma_(a0, bb0, c);
    c = wmma_(a1, bb1, c);
    float bv = bias[nb + (lane & 15)];
#pragma unroll
    for (int r = 0; r < 8; ++r) c[r] += bv;
    c_tile_store(xp + (mt * 32 + ntile) * 256, lane, c);
  }
}

// ---------------------------------------------------------------------------
// K2: sequential LSTM scan, one persistent workgroup per direction.
// Whh B-fragments live in 128 persistent VGPRs per wave; the xproj C-tile
// stream is software-pipelined one step ahead (loads for t+1 issue before the
// WMMAs of step t, draining across the barrier). Cell state c stays in
// registers for all 256 steps; h round-trips through 8 KB of LDS as f16.
__global__ void k_scan(const float* __restrict__ Whh_f, const float* __restrict__ Whh_b,
                       const float* __restrict__ xpf, const float* __restrict__ xpb,
                       float* __restrict__ hf, float* __restrict__ hbr) {
  extern __shared__ char smem[];
  _Float16* h16 = (_Float16*)smem;          // 32x128 f16 = 8 KB
  int dir = blockIdx.x;
  const float* Whh = dir ? Whh_b : Whh_f;
  const float* xp  = dir ? xpb : xpf;
  float* hout      = dir ? hbr : hf;
  int tid = threadIdx.x;
  for (int i = tid; i < 32 * 128; i += 256) h16[i] = (_Float16)0.0f;
  __syncthreads();
  int lane = tid & 31, w = tid >> 5;
  int eg = 16 * w + (lane & 15);
  int halfm = (lane >> 4) * 8;

  // one-time weight preload: 4 gates x 4 k-frags (128 persistent VGPRs)
  v16h wfr[4][4];
#pragma unroll
  for (int g = 0; g < 4; ++g)
#pragma unroll
    for (int kf = 0; kf < 4; ++kf)
      wfr[g][kf] = b_frag_w(Whh, 128, (w + 8 * g) * 16, kf * 32, lane);

  float creg[2][8];
#pragma unroll
  for (int mt = 0; mt < 2; ++mt)
#pragma unroll
    for (int r = 0; r < 8; ++r) creg[mt][r] = 0.0f;

  // prime the xproj tile pipeline with step 0
  v8f xpn[4][2];
#pragma unroll
  for (int g = 0; g < 4; ++g)
#pragma unroll
    for (int mt = 0; mt < 2; ++mt)
      xpn[g][mt] = c_tile_load(xp + ((mt * 32) + (w + 8 * g)) * 256, lane);

  for (int t = 0; t < 256; ++t) {
    v16h afr[2][4];
#pragma unroll
    for (int mt = 0; mt < 2; ++mt)
#pragma unroll
      for (int kf = 0; kf < 4; ++kf)
        afr[mt][kf] = a_frag_lds(h16, 128, mt * 16, kf * 32, lane);
    __syncthreads();                        // all waves read h before overwrite

    // consume pipelined tiles, then immediately issue loads for t+1
    v8f accs[4][2];
#pragma unroll
    for (int g = 0; g < 4; ++g)
#pragma unroll
      for (int mt = 0; mt < 2; ++mt) accs[g][mt] = xpn[g][mt];
    if (t < 255) {
#pragma unroll
      for (int g = 0; g < 4; ++g)
#pragma unroll
        for (int mt = 0; mt < 2; ++mt)
          xpn[g][mt] = c_tile_load(xp + ((((t + 1) * 2 + mt) * 32) + (w + 8 * g)) * 256, lane);
    }

#pragma unroll
    for (int g = 0; g < 4; ++g)
#pragma unroll
      for (int mt = 0; mt < 2; ++mt)
#pragma unroll
        for (int kf = 0; kf < 4; ++kf)
          accs[g][mt] = wmma_(afr[mt][kf], wfr[g][kf], accs[g][mt]);

#pragma unroll
    for (int mt = 0; mt < 2; ++mt)
#pragma unroll
      for (int r = 0; r < 8; ++r) {
        float iv = sigm_(accs[0][mt][r]);
        float fv = sigm_(accs[1][mt][r]);
        float gv = tanh_(accs[2][mt][r]);
        float ov = sigm_(accs[3][mt][r]);
        float cn = fv * creg[mt][r] + iv * gv;
        creg[mt][r] = cn;
        float h = ov * tanh_(cn);
        int b = mt * 16 + r + halfm;
        h16[b * 128 + eg] = (_Float16)h;
        hout[(t * 32 + b) * 128 + eg] = h;
      }
    __syncthreads();
  }
}

// ---------------------------------------------------------------------------
// K3a: enc = mask * [hf | un-reversed hb]; writes both [b][t][e] (for the
// enc_proj GEMM) and a transposed copy [b][e][t] so the decoder's per-step
// ctx reduction streams contiguously.
__global__ void k_enc(const float* __restrict__ hf, const float* __restrict__ hbr,
                      const int* __restrict__ len, const int* __restrict__ rev,
                      float* __restrict__ enc, float* __restrict__ encT) {
  int idx = blockIdx.x * 256 + threadIdx.x;   // B*T*256 threads
  int b = idx >> 16;
  int t = (idx >> 8) & 255;
  int e = idx & 255;
  float val = 0.0f;
  if (t < len[b]) {
    if (e < 128) val = hf[(t * 32 + b) * 128 + e];
    else {
      int tp = rev[b * 256 + t];
      val = hbr[(tp * 32 + b) * 128 + (e - 128)];
    }
  }
  enc[(b * 256 + t) * 256 + e] = val;
  encT[(b * 256 + e) * 256 + t] = val;
}

// ---------------------------------------------------------------------------
// K3b: enc_proj = enc @ Wa_e^T + ba   (M=8192, N=128, K=256); Wa_e = Wa[:,128:]
__global__ void k_encproj(const float* __restrict__ enc, const float* __restrict__ Wa,
                          const float* __restrict__ ba, float* __restrict__ encp) {
  int lane = threadIdx.x & 31, w = threadIdx.x >> 5;
  int mt = blockIdx.x;
  const float* rowp = enc + (mt * 16 + (lane & 15)) * 256;
  int nb = w * 16;                             // wave w -> n-tile w (8 tiles)
  v8f acc = {};
#pragma unroll
  for (int kf = 0; kf < 8; ++kf) {
    v16h a  = a_frag_row(rowp, kf * 32, lane);
    v16h bb = b_frag_w(Wa + 128, 384, nb, kf * 32, lane);   // skip Wa_h cols
    acc = wmma_(a, bb, acc);
  }
  float bv = ba[nb + (lane & 15)];
#pragma unroll
  for (int r = 0; r < 8; ++r) acc[r] += bv;
  c_frag_store(encp, 128, mt * 16, nb, lane, acc);
}

// ---------------------------------------------------------------------------
// K4: h0proj = dec_h0 @ Whh_d^T + b_d (loop-invariant in the reference decoder)
__global__ void k_h0proj(const float* __restrict__ dec_h0, const float* __restrict__ Whh_d,
                         const float* __restrict__ b_d, float* __restrict__ h0p) {
  int lane = threadIdx.x & 31, w = threadIdx.x >> 5;
#pragma unroll
  for (int nt = 0; nt < 4; ++nt) {
    int nb = (w * 4 + nt) * 16;
    v16h bf[4];
#pragma unroll
    for (int kf = 0; kf < 4; ++kf) bf[kf] = b_frag_w(Whh_d, 128, nb, kf * 32, lane);
    float bv = b_d[nb + (lane & 15)];
#pragma unroll
    for (int mt = 0; mt < 2; ++mt) {
      const float* rowp = dec_h0 + (mt * 16 + (lane & 15)) * 128;
      v8f acc = {};
#pragma unroll
      for (int kf = 0; kf < 4; ++kf) {
        v16h a = a_frag_row(rowp, kf * 32, lane);
        acc = wmma_(a, bf[kf], acc);
      }
#pragma unroll
      for (int r = 0; r < 8; ++r) acc[r] += bv;
      c_frag_store(h0p, 512, mt * 16, nb, lane, acc);
    }
  }
}

// ---------------------------------------------------------------------------
// K5: attention decoder, one persistent workgroup, 256 sequential steps.
// Loop-invariant operands (Wa_h frags, Wl frags + bias, h0proj C-frags,
// dec_c0 slice, lengths) preloaded into registers/LDS; the only per-step
// weight stream is Wih_d, read from a pre-converted f16 table; the ctx
// reduction streams the transposed encoder copy contiguously.
__global__ void k_decoder(const float* __restrict__ encT, const float* __restrict__ encp,
                          const float* __restrict__ h0p, const int* __restrict__ len,
                          const float* __restrict__ dec_h0, const float* __restrict__ dec_c0,
                          const float* __restrict__ Wa, const _Float16* __restrict__ wihd16,
                          const float* __restrict__ Wl, const float* __restrict__ bl,
                          const float* __restrict__ v, float* __restrict__ out) {
  extern __shared__ char smem[];
  _Float16* hidden16 = (_Float16*)smem;                // 32x128 f16   ( 8192 B)
  float*    aproj    = (float*)(smem + 8192);          // 32x128 f32   (16384 B)
  float*    s        = (float*)(smem + 24576);         // 32x256 f32   (32768 B)
  _Float16* di16     = (_Float16*)(smem + 57344);      // 32x288 f16   (18432 B)
  _Float16* hctx16   = (_Float16*)(smem + 75776);      // 32x384 f16   (24576 B)
  float*    yy       = (float*)(smem + 100352);        // 32x32  f32   ( 4096 B)
  float*    vv       = (float*)(smem + 104448);        // 128    f32   (  512 B)
  int*      lenS     = (int*)(smem + 104960);          // 32     i32   (  128 B)
  int tid = threadIdx.x;
  int lane = tid & 31, w = tid >> 5;
  int halfm = (lane >> 4) * 8;
  int eg = 16 * w + (lane & 15);

  for (int i = tid; i < 32 * 128; i += 256) hidden16[i] = (_Float16)dec_h0[i];
  for (int i = tid; i < 32 * 32; i += 256) yy[i] = 0.0f;
  if (tid < 128) vv[tid] = v[tid];
  if (tid < 32) lenS[tid] = len[tid];
  // output columns 0..30 are the reference's shift-by-one zeros
  for (int p = tid; p < 992; p += 256) out[(p / 31) * 8223 + (p % 31)] = 0.0f;

  // ---- loop-invariant register preloads ----
  v16h wa_f[4];
#pragma unroll
  for (int kf = 0; kf < 4; ++kf) wa_f[kf] = b_frag_w(Wa, 384, w * 16, kf * 32, lane);
  v8f h0_f[4][2];
#pragma unroll
  for (int g = 0; g < 4; ++g)
#pragma unroll
    for (int mt = 0; mt < 2; ++mt)
      h0_f[g][mt] = c_frag_load(h0p, 512, mt * 16, (w + 8 * g) * 16, lane);
  float c0r[2][8];
#pragma unroll
  for (int mt = 0; mt < 2; ++mt)
#pragma unroll
    for (int r = 0; r < 8; ++r)
      c0r[mt][r] = dec_c0[(mt * 16 + r + halfm) * 128 + eg];
  v16h wl_f[12];
  float blv;
  {
    int nt = w & 1;
#pragma unroll
    for (int kf = 0; kf < 12; ++kf) wl_f[kf] = b_frag_w(Wl, 384, nt * 16, kf * 32, lane);
    blv = bl[nt * 16 + (lane & 15)];
  }
  __syncthreads();

  for (int t = 0; t < 256; ++t) {
    // --- Phase A: aproj = hidden @ Wa_h^T (M=32, N=128, K=128) ---
    {
      int nb = w * 16;
#pragma unroll
      for (int mt = 0; mt < 2; ++mt) {
        v8f acc = {};
#pragma unroll
        for (int kf = 0; kf < 4; ++kf) {
          v16h a = a_frag_lds(hidden16, 128, mt * 16, kf * 32, lane);
          acc = wmma_(a, wa_f[kf], acc);
        }
        c_frag_store(aproj, 128, mt * 16, nb, lane, acc);
      }
    }
    __syncthreads();

    // --- Phase B: s[b][tt] = mask ? sum_d tanh(encp + aproj)*v : -inf ---
    for (int j = 0; j < 32; ++j) {
      int p = tid + 256 * j;
      int b = p >> 8, tt = p & 255;
      float sv;
      if (tt >= lenS[b]) sv = -3.0e38f;
      else {
        const float* ep = encp + (b * 256 + tt) * 128;
        const float* ap = aproj + b * 128;
        float acc = 0.0f;
#pragma unroll 4
        for (int d = 0; d < 128; ++d) acc += tanh_(ep[d] + ap[d]) * vv[d];
        sv = acc;
      }
      s[p] = sv;
    }
    __syncthreads();
    // masked softmax over t, one wave32 per 4 batch rows
#pragma unroll
    for (int i = 0; i < 4; ++i) {
      int b = w * 4 + i;
      float mx = -3.0e38f;
      for (int jj = lane; jj < 256; jj += 32) mx = fmaxf(mx, s[b * 256 + jj]);
#pragma unroll
      for (int d = 16; d > 0; d >>= 1) mx = fmaxf(mx, __shfl_xor(mx, d, 32));
      float sm = 0.0f;
      for (int jj = lane; jj < 256; jj += 32) {
        float e2 = __expf(s[b * 256 + jj] - mx);
        s[b * 256 + jj] = e2;
        sm += e2;
      }
#pragma unroll
      for (int d = 16; d > 0; d >>= 1) sm += __shfl_xor(sm, d, 32);
      float inv = 1.0f / sm;
      for (int jj = lane; jj < 256; jj += 32) s[b * 256 + jj] *= inv;
    }
    __syncthreads();

    // --- Phase C: ctx[b][e] = sum_t alpha*encT (contiguous); di = [y, ctx] ---
    for (int j = 0; j < 32; ++j) {
      int p = tid + 256 * j;
      int b = p >> 8, e = p & 255;
      const float* al = s + b * 256;
      const float* er = encT + (b * 256 + e) * 256;
      float acc = 0.0f;
#pragma unroll 4
      for (int tt = 0; tt < 256; ++tt) acc += al[tt] * er[tt];
      di16[b * 288 + 32 + e]    = (_Float16)acc;
      hctx16[b * 384 + 128 + e] = (_Float16)acc;
    }
#pragma unroll
    for (int j = 0; j < 4; ++j) {
      int q = tid + 256 * j;
      int b = q >> 5, o = q & 31;
      di16[b * 288 + o] = (_Float16)yy[b * 32 + o];
    }
    __syncthreads();

    // --- Phase D: z = di @ Wih_d^T + h0proj ; gate combine with dec_c0 ---
    {
      v16h afr[2][9];
#pragma unroll
      for (int mt = 0; mt < 2; ++mt)
#pragma unroll
        for (int kf = 0; kf < 9; ++kf)
          afr[mt][kf] = a_frag_lds(di16, 288, mt * 16, kf * 32, lane);
      v8f accs[4][2];
#pragma unroll
      for (int g = 0; g < 4; ++g) {
        int nb = (w + 8 * g) * 16;
        v16h bf[9];
#pragma unroll
        for (int kf = 0; kf < 9; ++kf) bf[kf] = b_frag_h16(wihd16, 288, nb, kf * 32, lane);
#pragma unroll
        for (int mt = 0; mt < 2; ++mt) {
          v8f acc = h0_f[g][mt];
#pragma unroll
          for (int kf = 0; kf < 9; ++kf) acc = wmma_(afr[mt][kf], bf[kf], acc);
          accs[g][mt] = acc;
        }
      }
#pragma unroll
      for (int mt = 0; mt < 2; ++mt)
#pragma unroll
        for (int r = 0; r < 8; ++r) {
          int b = mt * 16 + r + halfm;
          float iv = sigm_(accs[0][mt][r]);
          float fv = sigm_(accs[1][mt][r]);
          float gv = tanh_(accs[2][mt][r]);
          float ov = sigm_(accs[3][mt][r]);
          float cn = fv * c0r[mt][r] + iv * gv;
          float h = ov * tanh_(cn);
          hidden16[b * 128 + eg] = (_Float16)h;   // attention carry for t+1
          hctx16[b * 384 + eg]   = (_Float16)h;   // feeds y_new GEMM
        }
    }
    __syncthreads();

    // --- Phase E: y_new = [h, ctx] @ Wl^T + bl (M=32, N=32, K=384) ---
    if (w < 4) {
      int mt = w >> 1, nt = w & 1;
      v8f acc = {};
#pragma unroll
      for (int kf = 0; kf < 12; ++kf) {
        v16h a = a_frag_lds(hctx16, 384, mt * 16, kf * 32, lane);
        acc = wmma_(a, wl_f[kf], acc);
      }
      int n = nt * 16 + (lane & 15);
#pragma unroll
      for (int r = 0; r < 8; ++r) {
        int m = mt * 16 + r + halfm;
        float val = acc[r] + blv;
        yy[m * 32 + n] = val;                      // y carry for t+1
        out[m * 8223 + 31 + t * 32 + n] = val;     // shifted output layout
      }
    }
    __syncthreads();
  }
}

// ---------------------------------------------------------------------------
extern "C" void kernel_launch(void* const* d_in, const int* in_sizes, int n_in,
                              void* d_out, int out_size, void* d_ws, size_t ws_size,
                              hipStream_t stream) {
  const float* x      = (const float*)d_in[0];
  const int*   len    = (const int*)d_in[1];
  const float* dec_h0 = (const float*)d_in[2];
  const float* dec_c0 = (const float*)d_in[3];
  const float* Wih_f  = (const float*)d_in[4];
  const float* Whh_f  = (const float*)d_in[5];
  const float* b_f    = (const float*)d_in[6];
  const float* Wih_b  = (const float*)d_in[7];
  const float* Whh_b  = (const float*)d_in[8];
  const float* b_b    = (const float*)d_in[9];
  const float* Wa     = (const float*)d_in[10];
  const float* ba     = (const float*)d_in[11];
  const float* vvec   = (const float*)d_in[12];
  const float* Wih_d  = (const float*)d_in[13];
  const float* Whh_d  = (const float*)d_in[14];
  const float* b_d    = (const float*)d_in[15];
  const float* Wl     = (const float*)d_in[16];
  const float* bl     = (const float*)d_in[17];
  float* out = (float*)d_out;

  float* ws = (float*)d_ws;
  const size_t XPROJ = 256u * 32u * 512u;            // 4,194,304 floats each dir
  float* xpf  = ws;                                  // fragment-tiled
  float* xpb  = xpf + XPROJ;
  float* hf   = xpb + XPROJ;                         // [t][b][128]
  float* hbr  = hf + 256u * 32u * 128u;
  float* enc  = hbr + 256u * 32u * 128u;             // [b][t][256]
  float* encp = enc + 32u * 256u * 256u;             // [b][t][128]
  float* h0p  = encp + 32u * 256u * 128u;            // [32][512]
  int*   rev  = (int*)(h0p + 32u * 512u);            // [b][t]
  _Float16* wihd16 = (_Float16*)(rev + 32u * 256u);  // 512x288 f16
  float* encT = (float*)(wihd16 + 512u * 288u);      // [b][256e][256t]

  k_revidx <<<32, 256, 0, stream>>>(len, rev);
  k_cvt    <<<(512 * 288 + 255) / 256, 256, 0, stream>>>(Wih_d, wihd16, 512 * 288);
  k_xproj  <<<dim3(512, 2), 256, 0, stream>>>(x, Wih_f, b_f, Wih_b, b_b, rev, xpf, xpb);
  k_scan   <<<2, 256, 32 * 128 * 2, stream>>>(Whh_f, Whh_b, xpf, xpb, hf, hbr);
  k_enc    <<<8192, 256, 0, stream>>>(hf, hbr, len, rev, enc, encT);
  k_encproj<<<512, 256, 0, stream>>>(enc, Wa, ba, encp);
  k_h0proj <<<1, 256, 0, stream>>>(dec_h0, Whh_d, b_d, h0p);
  k_decoder<<<1, 256, 105088, stream>>>(encT, encp, h0p, len, dec_h0, dec_c0,
                                        Wa, wihd16, Wl, bl, vvec, out);
  (void)in_sizes; (void)n_in; (void)out_size; (void)ws_size;
}
